// SiteNetEncoder_2662879723993
// MI455X (gfx1250) — compile-verified, matrix operand order checked by reading, not run on hardware
//
#include <hip/hip_runtime.h>
#include <hip/hip_bf16.h>

typedef _Float16 h16;
typedef __attribute__((ext_vector_type(16))) _Float16 v16h;
typedef __attribute__((ext_vector_type(8)))  float v8f;

union Frag16 { v16h v; uint4 q[2]; };

#define DEV __device__ __forceinline__

// A-fragment (16x32 f16) from row-major f16 (LDS or global).
// Layout (CDNA5 ISA): lane m=lane&15, half h=lane>>4; element e: k = (e/8)*16 + h*8 + e%8.
DEV v16h load_a(const h16* base, int rowStride) {
  const int lane = threadIdx.x & 31;
  const h16* p = base + (lane & 15) * rowStride + ((lane >> 4) << 3);
  Frag16 f;
  f.q[0] = *(const uint4*)(p);
  f.q[1] = *(const uint4*)(p + 16);
  return f.v;
}

// B-fragment from pre-packed weights: tile = 32 lanes x 16 halves contiguous.
DEV v16h load_b(const h16* packedTile) {
  const int lane = threadIdx.x & 31;
  const uint4* p = (const uint4*)(packedTile + lane * 16);
  Frag16 f;
  f.q[0] = p[0];
  f.q[1] = p[1];
  return f.v;
}

DEV v8f wmma32(v16h a, v16h b, v8f c) {
  return __builtin_amdgcn_wmma_f32_16x16x32_f16(false, a, false, b, (short)0, c, false, false);
}

// CDNA5 async copy: 16 bytes global -> LDS, tracked by ASYNCcnt.
DEV void async_g2l_b128(const void* gptr, void* lptr) {
  unsigned lds = (unsigned)(uintptr_t)lptr;
  unsigned long long g = (unsigned long long)(uintptr_t)gptr;
  asm volatile("global_load_async_to_lds_b128 %0, %1, off" :: "v"(lds), "v"(g) : "memory");
}
DEV void async_wait0() {
  asm volatile("s_wait_asynccnt 0x0" ::: "memory");
}

// ---------------- weight packing: f32 [K,Nc] -> f16 WMMA-B fragment order ----
__global__ __launch_bounds__(256)
void pack_b_kernel(const float* __restrict__ W, int K, int Nc, h16* __restrict__ out, int total) {
  const int NT = Nc >> 4;
  for (int idx = blockIdx.x * blockDim.x + threadIdx.x; idx < total; idx += gridDim.x * blockDim.x) {
    int e    = idx & 15;
    int lane = (idx >> 4) & 31;
    int tile = idx >> 9;
    int nt = tile % NT, kt = tile / NT;
    int k = kt * 32 + ((e >> 3) << 4) + ((lane >> 4) << 3) + (e & 7);
    int c = nt * 16 + (lane & 15);
    out[idx] = (h16)W[(size_t)k * Nc + c];
  }
}

// ---------------- embeddings ------------------------------------------------
__global__ __launch_bounds__(256)
void site_embed_kernel(const float* __restrict__ SF, const float* __restrict__ W,
                       const float* __restrict__ b, h16* __restrict__ out) {
  const int n = blockIdx.x, c = threadIdx.x;
  const float* sf = SF + (size_t)n * 102;
  float acc = b[c];
  for (int k = 0; k < 102; ++k) acc += sf[k] * W[k * 256 + c];
  out[(size_t)n * 256 + c] = (h16)fmaxf(acc, 0.f);
}

__global__ __launch_bounds__(256)
void inter_embed_kernel(const float* __restrict__ IF, const float* __restrict__ W,
                        const float* __restrict__ b, h16* __restrict__ out) {
  const size_t r = blockIdx.x;
  const int c = threadIdx.x;
  const float* f = IF + r * 3;
  float acc = b[c] + f[0] * W[c] + f[1] * W[256 + c] + f[2] * W[512 + c];
  out[r * 256 + c] = (h16)fmaxf(acc, 0.f);
}

// ---------------- per-site attention block ----------------------------------
// One workgroup (8 wave32) per site n. x_ije = [site[n] | site[crystal] | inter[n]] in LDS.
// GEMMs are k-outer: each wave owns 2 n-tiles x 4 m-tiles (8 accumulators) so every
// packed-B fragment is loaded exactly once per workgroup.
__global__ __launch_bounds__(256)
void sitenet_block_kernel(
    const h16* __restrict__ site_in, int lda_site,
    h16* __restrict__ inter,                 // [N,64,256] f16, in/out (in-place carry)
    const unsigned char* __restrict__ mask,  // [N,64] bool
    const h16* __restrict__ pWm1, const float* __restrict__ bm1,
    const h16* __restrict__ pWm2, const float* __restrict__ bm2,
    const float* __restrict__ Wps, const float* __restrict__ bps,
    const h16* __restrict__ pWatt, const float* __restrict__ batt,
    const h16* __restrict__ pWif, const float* __restrict__ bif,
    h16* __restrict__ agg_out)               // [N,256] pre-gate
{
  __shared__ h16 s_site[64 * 256];   // crystal site rows (32 KB)
  __shared__ h16 s_inter[64 * 256];  // inter rows of site n (32 KB)
  __shared__ h16 s_h1[64 * 256];
  __shared__ h16 s_h2[64 * 256];
  __shared__ h16 s_feat[64 * 64];
  __shared__ float s_logit[64 * 4];
  __shared__ float s_w[64 * 4];

  const int n = blockIdx.x;
  const int cb = n & ~63;     // crystal base site
  const int ri = n & 63;      // row of site n within crystal
  const int t = threadIdx.x;
  const int wave = t >> 5, lane = t & 31;

  // async stage: crystal site tile + inter tile (2048 x b128 each)
  for (int i = t; i < 2048; i += 256) {
    int row = i >> 5, c8 = i & 31;
    async_g2l_b128(site_in + (size_t)(cb + row) * lda_site + c8 * 8,
                   s_site + row * 256 + c8 * 8);
    async_g2l_b128(inter + (size_t)n * 16384 + i * 8, s_inter + i * 8);
  }
  async_wait0();
  __syncthreads();

  // x_ije A-fragment regions: k [0,256)=site[n] (const row), [256,512)=site[j], [512,768)=inter
#define XIJE_BASE(kt) ((kt) < 8 ? s_site + ri * 256 + (kt) * 32 \
                      : (kt) < 16 ? s_site + ((kt) - 8) * 32    \
                                  : s_inter + ((kt) - 16) * 32)

  const int nt0 = wave * 2, nt1 = nt0 + 1;
  const int cl = lane & 15, rh = (lane >> 4) << 3;

  // GEMM1: h1 = relu(x_ije @ Wm1 + bm1), [64,768]@[768,256]
  {
    v8f acc[4][2] = {};
    for (int kt = 0; kt < 24; ++kt) {
      v16h b0 = load_b(pWm1 + (size_t)(kt * 16 + nt0) * 512);
      v16h b1 = load_b(pWm1 + (size_t)(kt * 16 + nt1) * 512);
      const h16* base = XIJE_BASE(kt);
      if (kt < 8) {
        v16h a = load_a(base, 0);
        for (int mt = 0; mt < 4; ++mt) {
          acc[mt][0] = wmma32(a, b0, acc[mt][0]);
          acc[mt][1] = wmma32(a, b1, acc[mt][1]);
        }
      } else {
        for (int mt = 0; mt < 4; ++mt) {
          v16h a = load_a(base + mt * 16 * 256, 256);
          acc[mt][0] = wmma32(a, b0, acc[mt][0]);
          acc[mt][1] = wmma32(a, b1, acc[mt][1]);
        }
      }
    }
    for (int mt = 0; mt < 4; ++mt)
      for (int j = 0; j < 2; ++j) {
        int nt = nt0 + j;
        float bv = bm1[nt * 16 + cl];
        for (int e = 0; e < 8; ++e)
          s_h1[(mt * 16 + rh + e) * 256 + nt * 16 + cl] = (h16)fmaxf(acc[mt][j][e] + bv, 0.f);
      }
  }
  __syncthreads();

  // GEMM2: h2 = relu(h1 @ Wm2 + bm2), [64,256]@[256,256]
  {
    v8f acc[4][2] = {};
    for (int kt = 0; kt < 8; ++kt) {
      v16h b0 = load_b(pWm2 + (size_t)(kt * 16 + nt0) * 512);
      v16h b1 = load_b(pWm2 + (size_t)(kt * 16 + nt1) * 512);
      for (int mt = 0; mt < 4; ++mt) {
        v16h a = load_a(s_h1 + mt * 16 * 256 + kt * 32, 256);
        acc[mt][0] = wmma32(a, b0, acc[mt][0]);
        acc[mt][1] = wmma32(a, b1, acc[mt][1]);
      }
    }
    for (int mt = 0; mt < 4; ++mt)
      for (int j = 0; j < 2; ++j) {
        int nt = nt0 + j;
        float bv = bm2[nt * 16 + cl];
        for (int e = 0; e < 8; ++e)
          s_h2[(mt * 16 + rh + e) * 256 + nt * 16 + cl] = (h16)fmaxf(acc[mt][j][e] + bv, 0.f);
      }
  }
  __syncthreads();

  // logits = h2 @ Wps + bps, mask -> -inf   (VALU: 256 threads = (m,h) pairs)
  {
    int m = t >> 2, hh = t & 3;
    float acc = bps[hh];
    for (int k = 0; k < 256; ++k) acc += (float)s_h2[m * 256 + k] * Wps[k * 4 + hh];
    if (mask[(size_t)n * 64 + m]) acc = -__builtin_inff();
    s_logit[m * 4 + hh] = acc;
  }
  __syncthreads();

  // top-5 softmax over m per head
  if (t < 4) {
    int hh = t;
    unsigned long long sel = 0ull;
    for (int it = 0; it < 5; ++it) {
      float best = -__builtin_inff(); int bi = -1;
      for (int m = 0; m < 64; ++m) {
        if ((sel >> m) & 1ull) continue;
        float v = s_logit[m * 4 + hh];
        if (v > best) { best = v; bi = m; }
      }
      if (bi >= 0) sel |= 1ull << bi;
    }
    float mx = -__builtin_inff();
    for (int m = 0; m < 64; ++m) if ((sel >> m) & 1ull) mx = fmaxf(mx, s_logit[m * 4 + hh]);
    float ssum = 0.f;
    for (int m = 0; m < 64; ++m) {
      float e = ((sel >> m) & 1ull) ? __expf(s_logit[m * 4 + hh] - mx) : 0.f;
      s_w[m * 4 + hh] = e; ssum += e;
    }
    float inv = 1.f / ssum;
    for (int m = 0; m < 64; ++m) s_w[m * 4 + hh] *= inv;
  }
  __syncthreads();

  // feat = relu(x_ije @ Watt + batt), [64,768]@[768,64]
  // wave -> n-tile (wave&3), m-tiles {2*(wave>>2), +1}: all 8 waves active.
  {
    const int nt = wave & 3;
    const int mtb = (wave >> 2) * 2;
    v8f acc[2] = {};
    for (int kt = 0; kt < 24; ++kt) {
      v16h b = load_b(pWatt + (size_t)(kt * 4 + nt) * 512);
      const h16* base = XIJE_BASE(kt);
      if (kt < 8) {
        v16h a = load_a(base, 0);
        acc[0] = wmma32(a, b, acc[0]);
        acc[1] = wmma32(a, b, acc[1]);
      } else {
        for (int q = 0; q < 2; ++q) {
          v16h a = load_a(base + (mtb + q) * 16 * 256, 256);
          acc[q] = wmma32(a, b, acc[q]);
        }
      }
    }
    float bv = batt[nt * 16 + cl];
    for (int q = 0; q < 2; ++q)
      for (int e = 0; e < 8; ++e)
        s_feat[((mtb + q) * 16 + rh + e) * 64 + nt * 16 + cl] = (h16)fmaxf(acc[q][e] + bv, 0.f);
  }
  __syncthreads();

  // agg[n, e*4+h] = sum_m w[m,h] * feat[m,e]
  {
    int e = t >> 2, hh = t & 3;
    float acc = 0.f;
    for (int m = 0; m < 64; ++m) acc += s_w[m * 4 + hh] * (float)s_feat[m * 64 + e];
    agg_out[(size_t)n * 256 + e * 4 + hh] = (h16)acc;
  }

  // new_inter = relu(x_ije @ Wif + bif) -> global inter[n] (in place)
  {
    v8f acc[4][2] = {};
    for (int kt = 0; kt < 24; ++kt) {
      v16h b0 = load_b(pWif + (size_t)(kt * 16 + nt0) * 512);
      v16h b1 = load_b(pWif + (size_t)(kt * 16 + nt1) * 512);
      const h16* base = XIJE_BASE(kt);
      if (kt < 8) {
        v16h a = load_a(base, 0);
        for (int mt = 0; mt < 4; ++mt) {
          acc[mt][0] = wmma32(a, b0, acc[mt][0]);
          acc[mt][1] = wmma32(a, b1, acc[mt][1]);
        }
      } else {
        for (int mt = 0; mt < 4; ++mt) {
          v16h a = load_a(base + mt * 16 * 256, 256);
          acc[mt][0] = wmma32(a, b0, acc[mt][0]);
          acc[mt][1] = wmma32(a, b1, acc[mt][1]);
        }
      }
    }
    for (int mt = 0; mt < 4; ++mt)
      for (int j = 0; j < 2; ++j) {
        int nt = nt0 + j;
        float bv = bif[nt * 16 + cl];
        for (int e = 0; e < 8; ++e)
          inter[((size_t)n * 64 + mt * 16 + rh + e) * 256 + nt * 16 + cl] =
              (h16)fmaxf(acc[mt][j][e] + bv, 0.f);
      }
  }
#undef XIJE_BASE
}

// ---------------- generic 64-row x 256-col WMMA GEMM: relu(A @ Bp + bias) ---
__global__ __launch_bounds__(256)
void gemm64_kernel(const h16* __restrict__ A, int lda, int nkt,
                   const h16* __restrict__ Bp,
                   const float* __restrict__ bias,
                   h16* __restrict__ out16, float* __restrict__ out32,
                   int ldo, int ocol, int do_relu)
{
  const int row0 = blockIdx.x * 64;
  const int t = threadIdx.x, wave = t >> 5, lane = t & 31;
  const int nt0 = wave * 2, nt1 = nt0 + 1;
  v8f acc[4][2] = {};
  for (int kt = 0; kt < nkt; ++kt) {
    v16h b0 = load_b(Bp + (size_t)(kt * 16 + nt0) * 512);
    v16h b1 = load_b(Bp + (size_t)(kt * 16 + nt1) * 512);
    for (int mt = 0; mt < 4; ++mt) {
      v16h a = load_a(A + (size_t)(row0 + mt * 16) * lda + kt * 32, lda);
      acc[mt][0] = wmma32(a, b0, acc[mt][0]);
      acc[mt][1] = wmma32(a, b1, acc[mt][1]);
    }
  }
  const int cl = lane & 15, rh = (lane >> 4) << 3;
  for (int mt = 0; mt < 4; ++mt)
    for (int j = 0; j < 2; ++j) {
      int nt = nt0 + j;
      float bv = bias[nt * 16 + cl];
      for (int e = 0; e < 8; ++e) {
        float v = acc[mt][j][e] + bv;
        if (do_relu) v = fmaxf(v, 0.f);
        size_t r = row0 + mt * 16 + rh + e;
        int c = ocol + nt * 16 + cl;
        if (out32) out32[r * ldo + c] = v;
        else       out16[r * ldo + c] = (h16)v;
      }
    }
}

// ---------------- segment-mean pool + post MLP (one workgroup) --------------
__global__ __launch_bounds__(256)
void pool_post_kernel(const float* __restrict__ g2,
                      const h16* __restrict__ pW1, const float* __restrict__ b1,
                      const h16* __restrict__ pW2, const float* __restrict__ b2,
                      float* __restrict__ out)
{
  __shared__ h16 s_pool[32 * 256];
  __shared__ h16 s_h[32 * 256];
  const int t = threadIdx.x, wave = t >> 5, lane = t & 31;
  const int nt0 = wave * 2, nt1 = nt0 + 1;
  const int cl = lane & 15, rh = (lane >> 4) << 3;
  for (int idx = t; idx < 32 * 256; idx += 256) {
    int bcr = idx >> 8, c = idx & 255;
    float s = 0.f;
    for (int srow = 0; srow < 64; ++srow) s += g2[(size_t)(bcr * 64 + srow) * 256 + c];
    s_pool[idx] = (h16)(s * (1.f / 64.f));
  }
  __syncthreads();
  {
    v8f acc[2][2] = {};
    for (int kt = 0; kt < 8; ++kt) {
      v16h b0 = load_b(pW1 + (size_t)(kt * 16 + nt0) * 512);
      v16h b1 = load_b(pW1 + (size_t)(kt * 16 + nt1) * 512);
      for (int mt = 0; mt < 2; ++mt) {
        v16h a = load_a(s_pool + mt * 16 * 256 + kt * 32, 256);
        acc[mt][0] = wmma32(a, b0, acc[mt][0]);
        acc[mt][1] = wmma32(a, b1, acc[mt][1]);
      }
    }
    for (int mt = 0; mt < 2; ++mt)
      for (int j = 0; j < 2; ++j) {
        int nt = nt0 + j;
        float bv = b1[nt * 16 + cl];
        for (int e = 0; e < 8; ++e)
          s_h[(mt * 16 + rh + e) * 256 + nt * 16 + cl] = (h16)fmaxf(acc[mt][j][e] + bv, 0.f);
      }
  }
  __syncthreads();
  {
    v8f acc[2][2] = {};
    for (int kt = 0; kt < 8; ++kt) {
      v16h b0 = load_b(pW2 + (size_t)(kt * 16 + nt0) * 512);
      v16h b1 = load_b(pW2 + (size_t)(kt * 16 + nt1) * 512);
      for (int mt = 0; mt < 2; ++mt) {
        v16h a = load_a(s_h + mt * 16 * 256 + kt * 32, 256);
        acc[mt][0] = wmma32(a, b0, acc[mt][0]);
        acc[mt][1] = wmma32(a, b1, acc[mt][1]);
      }
    }
    for (int mt = 0; mt < 2; ++mt)
      for (int j = 0; j < 2; ++j) {
        int nt = nt0 + j;
        float bv = b2[nt * 16 + cl];
        for (int e = 0; e < 8; ++e)
          out[(size_t)(mt * 16 + rh + e) * 256 + nt * 16 + cl] = fmaxf(acc[mt][j][e] + bv, 0.f);
      }
  }
}

// ---------------- host orchestration ----------------------------------------
extern "C" void kernel_launch(void* const* d_in, const int* in_sizes, int n_in,
                              void* d_out, int out_size, void* d_ws, size_t ws_size,
                              hipStream_t stream)
{
  const int N = 2048, M = 64, L = 4;
  const float* SF     = (const float*)d_in[0];
  const float* IF     = (const float*)d_in[1];
  const unsigned char* Mask = (const unsigned char*)d_in[4];
  const float* W_site = (const float*)d_in[6];
  const float* b_site = (const float*)d_in[7];
  const float* W_if   = (const float*)d_in[8];
  const float* b_if   = (const float*)d_in[9];
  const float* Wm1  = (const float*)d_in[10];
  const float* bm1  = (const float*)d_in[11];
  const float* Wm2  = (const float*)d_in[12];
  const float* bm2  = (const float*)d_in[13];
  const float* Wps  = (const float*)d_in[14];
  const float* bps  = (const float*)d_in[15];
  const float* Watt = (const float*)d_in[16];
  const float* batt = (const float*)d_in[17];
  const float* Wif  = (const float*)d_in[18];
  const float* bif  = (const float*)d_in[19];
  const float* Wg   = (const float*)d_in[20];
  const float* bg   = (const float*)d_in[21];
  const float* Wpre1  = (const float*)d_in[22];
  const float* bpre1  = (const float*)d_in[23];
  const float* Wpre2  = (const float*)d_in[24];
  const float* bpre2  = (const float*)d_in[25];
  const float* Wpost1 = (const float*)d_in[26];
  const float* bpost1 = (const float*)d_in[27];
  const float* Wpost2 = (const float*)d_in[28];
  const float* bpost2 = (const float*)d_in[29];

  char* ws = (char*)d_ws;
  size_t off = 0;
  auto alloc = [&](size_t bytes) -> char* {
    char* p = ws + off;
    off = (off + bytes + 255) & ~(size_t)255;
    return p;
  };
  h16* site16  = (h16*)alloc((size_t)N * 256 * 2);
  h16* inter16 = (h16*)alloc((size_t)N * M * 256 * 2);
  h16* agg16   = (h16*)alloc((size_t)N * 256 * 2);
  h16* xcat16  = (h16*)alloc((size_t)N * 1024 * 2);
  h16* g1_16   = (h16*)alloc((size_t)N * 256 * 2);
  float* g2_32 = (float*)alloc((size_t)N * 256 * 4);
  h16* pWm1  = (h16*)alloc((size_t)L * 768 * 256 * 2);
  h16* pWm2  = (h16*)alloc((size_t)L * 256 * 256 * 2);
  h16* pWatt = (h16*)alloc((size_t)L * 768 * 64 * 2);
  h16* pWif  = (h16*)alloc((size_t)L * 768 * 256 * 2);
  h16* pWg   = (h16*)alloc((size_t)L * 256 * 256 * 2);
  h16* pWpre1  = (h16*)alloc((size_t)1024 * 256 * 2);
  h16* pWpre2  = (h16*)alloc((size_t)256 * 256 * 2);
  h16* pWpost1 = (h16*)alloc((size_t)256 * 256 * 2);
  h16* pWpost2 = (h16*)alloc((size_t)256 * 256 * 2);

  auto pack = [&](const float* W, int K, int Nc, h16* out) {
    int total = K * Nc;
    int blocks = (total + 255) / 256;
    if (blocks > 2048) blocks = 2048;
    pack_b_kernel<<<blocks, 256, 0, stream>>>(W, K, Nc, out, total);
  };
  for (int l = 0; l < L; ++l) {
    pack(Wm1  + (size_t)l * 768 * 256, 768, 256, pWm1  + (size_t)l * 768 * 256);
    pack(Wm2  + (size_t)l * 256 * 256, 256, 256, pWm2  + (size_t)l * 256 * 256);
    pack(Watt + (size_t)l * 768 * 64,  768, 64,  pWatt + (size_t)l * 768 * 64);
    pack(Wif  + (size_t)l * 768 * 256, 768, 256, pWif  + (size_t)l * 768 * 256);
    pack(Wg   + (size_t)l * 256 * 256, 256, 256, pWg   + (size_t)l * 256 * 256);
  }
  pack(Wpre1, 1024, 256, pWpre1);
  pack(Wpre2, 256, 256, pWpre2);
  pack(Wpost1, 256, 256, pWpost1);
  pack(Wpost2, 256, 256, pWpost2);

  site_embed_kernel<<<N, 256, 0, stream>>>(SF, W_site, b_site, site16);
  inter_embed_kernel<<<N * M, 256, 0, stream>>>(IF, W_if, b_if, inter16);

  for (int l = 0; l < L; ++l) {
    const h16* sin = (l == 0) ? site16 : (xcat16 + (size_t)(l - 1) * 256);
    int lda = (l == 0) ? 256 : 1024;
    sitenet_block_kernel<<<N, 256, 0, stream>>>(
        sin, lda, inter16, Mask,
        pWm1  + (size_t)l * 768 * 256, bm1 + l * 256,
        pWm2  + (size_t)l * 256 * 256, bm2 + l * 256,
        Wps   + (size_t)l * 256 * 4,   bps + l * 4,
        pWatt + (size_t)l * 768 * 64,  batt + l * 64,
        pWif  + (size_t)l * 768 * 256, bif + l * 256,
        agg16);
    // gate: xs[l] = relu(agg @ Wg + bg) -> concat buffer cols [l*256, l*256+256)
    gemm64_kernel<<<N / 64, 256, 0, stream>>>(agg16, 256, 8,
        pWg + (size_t)l * 256 * 256, bg + l * 256,
        xcat16, nullptr, 1024, l * 256, 1);
  }

  gemm64_kernel<<<N / 64, 256, 0, stream>>>(xcat16, 1024, 32, pWpre1, bpre1,
                                            g1_16, nullptr, 256, 0, 1);
  gemm64_kernel<<<N / 64, 256, 0, stream>>>(g1_16, 256, 8, pWpre2, bpre2,
                                            nullptr, g2_32, 256, 0, 1);
  pool_post_kernel<<<1, 256, 0, stream>>>(g2_32, pWpost1, bpost1, pWpost2, bpost2,
                                          (float*)d_out);
}
